// probattn_block_575525618091
// MI455X (gfx1250) — matrix operand order repaired and checked by reading, not compiled, not run on hardware
//
#include <hip/hip_runtime.h>
#include <hip/hip_bf16.h>
#include <math.h>

typedef __attribute__((ext_vector_type(16))) __bf16 v16bf;
typedef __attribute__((ext_vector_type(8)))  float  v8f;

#define EPSF 1e-5f

// ---------------------------------------------------------------------------
// Phase A: attention helper kernels (latency-bound, f32 VALU)
// Attention tensor layout: T[a][h][s][d] flat = ((a*8+h)*512 + s)*4 + d
//   a in [0,128) (was L), h in [0,8), s in [0,512) (was B), d in [0,4)
// ---------------------------------------------------------------------------

__global__ void qkv_kernel(const float* __restrict__ x,
                           const float* __restrict__ Wq, const float* __restrict__ bq,
                           const float* __restrict__ Wk, const float* __restrict__ bk,
                           const float* __restrict__ Wv, const float* __restrict__ bv,
                           float* __restrict__ Q, float* __restrict__ K,
                           float* __restrict__ V) {
  int bid = blockIdx.x;            // a*512 + s
  int a = bid >> 9;
  int s = bid & 511;
  int t = threadIdx.x;             // output channel 0..31
  __shared__ float xv[32];
  xv[t] = x[(size_t)s * 4096 + t * 128 + a];   // x[b=s][c=t][l=a]
  __syncthreads();
  float aq = bq[t], ak = bk[t], av = bv[t];
#pragma unroll
  for (int c = 0; c < 32; ++c) {
    float xc = xv[c];
    aq += Wq[t * 32 + c] * xc;
    ak += Wk[t * 32 + c] * xc;
    av += Wv[t * 32 + c] * xc;
  }
  int h = t >> 2, d = t & 3;
  size_t o = ((size_t)((a * 8 + h) * 512 + s)) * 4 + d;
  Q[o] = aq; K[o] = ak; V[o] = av;
}

// M[a,h,s] = max_u(Q.K_sample) - sum_u(Q.K_sample)/512
__global__ void msample_kernel(const float* __restrict__ Q,
                               const float* __restrict__ Km,
                               const int* __restrict__ idx,
                               float* __restrict__ M) {
  int tid = blockIdx.x * blockDim.x + threadIdx.x;
  if (tid >= 128 * 8 * 512) return;
  int s  = tid & 511;
  int ah = tid >> 9;
  const float4* Q4 = (const float4*)Q;
  const float4* K4 = (const float4*)Km;
  float4 q = Q4[(size_t)ah * 512 + s];
  float mx = -INFINITY, sum = 0.f;
  for (int u = 0; u < 35; ++u) {
    int j = idx[s * 35 + u];
    float4 kk = K4[(size_t)ah * 512 + j];
    float d = q.x * kk.x + q.y * kk.y + q.z * kk.z + q.w * kk.w;
    mx = fmaxf(mx, d);
    sum += d;
  }
  M[tid] = mx - sum * (1.0f / 512.0f);
}

// top-35 indices of M per (a,h); ties -> lower index (matches lax.top_k)
__global__ void topk_kernel(const float* __restrict__ M, int* __restrict__ Mtop) {
  int ah = blockIdx.x;             // 0..1023
  int t  = threadIdx.x;            // 0..255
  __shared__ float sv[512];
  __shared__ float rv[256];
  __shared__ int   ri[256];
  sv[t]       = M[(size_t)ah * 512 + t];
  sv[t + 256] = M[(size_t)ah * 512 + t + 256];
  __syncthreads();
  for (int it = 0; it < 35; ++it) {
    float v0 = sv[t]; int i0 = t;
    float v1 = sv[t + 256];
    if (v1 > v0) { v0 = v1; i0 = t + 256; }
    rv[t] = v0; ri[t] = i0;
    __syncthreads();
    for (int off = 128; off > 0; off >>= 1) {
      if (t < off) {
        float vb = rv[t + off]; int ib = ri[t + off];
        if (vb > rv[t] || (vb == rv[t] && ib < ri[t])) { rv[t] = vb; ri[t] = ib; }
      }
      __syncthreads();
    }
    if (t == 0) {
      Mtop[ah * 35 + it] = ri[0];
      sv[ri[0]] = -INFINITY;
    }
    __syncthreads();
  }
}

// causal-masked softmax over all 512 keys for each selected query -> upd
__global__ void attn_kernel(const float* __restrict__ Q,
                            const float* __restrict__ Km,
                            const float* __restrict__ V,
                            const int* __restrict__ Mtop,
                            float* __restrict__ upd) {
  int bid = blockIdx.x;
  int u  = bid % 35;
  int ah = bid / 35;
  int t  = threadIdx.x;            // 0..127
  int jtop = Mtop[ah * 35 + u];
  const float4* Q4 = (const float4*)Q;
  const float4* K4 = (const float4*)Km;
  const float4* V4 = (const float4*)V;
  float4 q = Q4[(size_t)ah * 512 + jtop];
  float sc[4];
  float lmax = -INFINITY;
#pragma unroll
  for (int i = 0; i < 4; ++i) {
    int k = t + i * 128;
    float4 kk = K4[(size_t)ah * 512 + k];
    float d = 0.5f * (q.x * kk.x + q.y * kk.y + q.z * kk.z + q.w * kk.w);
    sc[i] = d;
    if (k <= jtop) lmax = fmaxf(lmax, d);
  }
  __shared__ float red[128];
  red[t] = lmax;
  __syncthreads();
  for (int off = 64; off > 0; off >>= 1) {
    if (t < off) red[t] = fmaxf(red[t], red[t + off]);
    __syncthreads();
  }
  float mx = red[0];
  __syncthreads();
  float esum = 0.f, w0 = 0.f, w1 = 0.f, w2 = 0.f, w3 = 0.f;
#pragma unroll
  for (int i = 0; i < 4; ++i) {
    int k = t + i * 128;
    if (k <= jtop) {
      float e = expf(sc[i] - mx);
      esum += e;
      float4 vv = V4[(size_t)ah * 512 + k];
      w0 += e * vv.x; w1 += e * vv.y; w2 += e * vv.z; w3 += e * vv.w;
    }
  }
  __shared__ float racc[128 * 5];
  racc[t * 5 + 0] = esum; racc[t * 5 + 1] = w0; racc[t * 5 + 2] = w1;
  racc[t * 5 + 3] = w2;   racc[t * 5 + 4] = w3;
  __syncthreads();
  for (int off = 64; off > 0; off >>= 1) {
    if (t < off) {
#pragma unroll
      for (int j = 0; j < 5; ++j) racc[t * 5 + j] += racc[(t + off) * 5 + j];
    }
    __syncthreads();
  }
  if (t == 0) {
    float inv = 1.0f / racc[0];
    float4 r;
    r.x = racc[1] * inv; r.y = racc[2] * inv; r.z = racc[3] * inv; r.w = racc[4] * inv;
    ((float4*)upd)[(size_t)ah * 35 + u] = r;
  }
}

// ctx = inclusive cumsum of V over s (Hillis-Steele on 512 float4)
__global__ void cumsum_kernel(const float* __restrict__ V, float* __restrict__ ctx) {
  int ah = blockIdx.x;
  int s  = threadIdx.x;            // 0..511
  __shared__ float4 sv[512];
  sv[s] = ((const float4*)V)[(size_t)ah * 512 + s];
  __syncthreads();
  for (int off = 1; off < 512; off <<= 1) {
    float4 add = {0.f, 0.f, 0.f, 0.f};
    if (s >= off) add = sv[s - off];
    __syncthreads();
    sv[s].x += add.x; sv[s].y += add.y; sv[s].z += add.z; sv[s].w += add.w;
    __syncthreads();
  }
  ((float4*)ctx)[(size_t)ah * 512 + s] = sv[s];
}

__global__ void scatter_kernel(const float* __restrict__ upd,
                               const int* __restrict__ Mtop,
                               float* __restrict__ ctx) {
  int tid = blockIdx.x * blockDim.x + threadIdx.x;
  if (tid >= 1024 * 35) return;
  int u  = tid % 35;
  int ah = tid / 35;
  int j = Mtop[ah * 35 + u];
  ((float4*)ctx)[(size_t)ah * 512 + j] = ((const float4*)upd)[(size_t)ah * 35 + u];
}

// out = relu(ctx @ Wo^T + bo); xh = BN1(out + x), stored flat [b, c*128 + l]
__global__ void outproj_bn1_kernel(const float* __restrict__ ctx,
                                   const float* __restrict__ Wo, const float* __restrict__ bo,
                                   const float* __restrict__ x,
                                   const float* __restrict__ g1, const float* __restrict__ be1,
                                   const float* __restrict__ m1, const float* __restrict__ v1,
                                   float* __restrict__ xh) {
  int bid = blockIdx.x;            // a*512 + s
  int a = bid >> 9;
  int s = bid & 511;
  int t = threadIdx.x;             // channel 0..31
  __shared__ float cv[32];
  cv[t] = ctx[((size_t)((a * 8 + (t >> 2)) * 512 + s)) * 4 + (t & 3)];
  __syncthreads();
  float o = bo[t];
#pragma unroll
  for (int j = 0; j < 32; ++j) o += Wo[t * 32 + j] * cv[j];
  o = fmaxf(o, 0.f);
  o += x[(size_t)s * 4096 + t * 128 + a];                 // residual res1
  o = (o - m1[t]) * rsqrtf(v1[t] + EPSF) * g1[t] + be1[t];
  xh[(size_t)s * 4096 + t * 128 + a] = o;
}

// ---------------------------------------------------------------------------
// Phase B/C: bf16 WMMA GEMM  Y[m,n] = epilogue( sum_k X[m,k]*W[n,k] + bias[n] )
//   WG: 256 threads (8 waves), computes 128(M) x 128(N); wave = 64x32 (4x2 tiles)
//   A frag (16x32): lane m -> K {kh*8..kh*8+7} and {16+kh*8..16+kh*8+7}
//   B frag (32x16): lane n -> K {kh*8..kh*8+7} and {16+kh*8..16+kh*8+7}   (kh = lane>>4)
//   LAYER==1: A = f32 (cvt inline), store Y as bf16, ReLU only.
//   LAYER==2: A = bf16 (raw b128 loads, no cvt), store f32, ReLU+resid+BN2.
//   Explicit 2-stage software pipeline: fragment buffers A0/B0 and A1/B1
//   ping-pong so cvts never WAR an in-flight WMMA's source registers.
// ---------------------------------------------------------------------------

__device__ inline v16bf load_a_frag_f32(const float* __restrict__ row, int k0, int kh) {
  const float4* p0 = (const float4*)(row + k0 + kh * 8);
  const float4* p1 = (const float4*)(row + k0 + 16 + kh * 8);
  float4 c0 = p0[0], c1 = p0[1], c2 = p1[0], c3 = p1[1];
  float f[16] = {c0.x, c0.y, c0.z, c0.w, c1.x, c1.y, c1.z, c1.w,
                 c2.x, c2.y, c2.z, c2.w, c3.x, c3.y, c3.z, c3.w};
  v16bf r;
#pragma unroll
  for (int i = 0; i < 16; ++i) r[i] = (__bf16)f[i];
  return r;
}

__device__ inline v16bf load_a_frag_bf16(const __bf16* __restrict__ row, int k0, int kh) {
  union { uint4 u[2]; v16bf v; } t;
  t.u[0] = *(const uint4*)(row + k0 + kh * 8);        // 8 bf16 = 16B
  t.u[1] = *(const uint4*)(row + k0 + 16 + kh * 8);   // 8 bf16 = 16B
  return t.v;
}

__device__ inline v16bf load_b_frag(const float* __restrict__ row, int k0, int kh) {
  const float4* p0 = (const float4*)(row + k0 + kh * 8);
  const float4* p1 = (const float4*)(row + k0 + 16 + kh * 8);
  float4 c0 = p0[0], c1 = p0[1], c2 = p1[0], c3 = p1[1];
  float f[16] = {c0.x, c0.y, c0.z, c0.w, c1.x, c1.y, c1.z, c1.w,
                 c2.x, c2.y, c2.z, c2.w, c3.x, c3.y, c3.z, c3.w};
  v16bf r;
#pragma unroll
  for (int i = 0; i < 16; ++i) r[i] = (__bf16)f[i];
  return r;
}

template <int LAYER>
__device__ inline void load_frags(v16bf a[4], v16bf b[2],
                                  const float* const arf[4],
                                  const __bf16* const arb[4],
                                  const float* const br[2],
                                  int k0, int kh) {
#pragma unroll
  for (int mt = 0; mt < 4; ++mt)
    a[mt] = (LAYER == 1) ? load_a_frag_f32(arf[mt], k0, kh)
                         : load_a_frag_bf16(arb[mt], k0, kh);
  b[0] = load_b_frag(br[0], k0, kh);
  b[1] = load_b_frag(br[1], k0, kh);
}

__device__ inline void wmma8(v8f acc[4][2], const v16bf a[4], const v16bf b[2]) {
#pragma unroll
  for (int mt = 0; mt < 4; ++mt) {
    acc[mt][0] = __builtin_amdgcn_wmma_f32_16x16x32_bf16(
        false, a[mt], false, b[0], (short)0, acc[mt][0], false, false);
    acc[mt][1] = __builtin_amdgcn_wmma_f32_16x16x32_bf16(
        false, a[mt], false, b[1], (short)0, acc[mt][1], false, false);
  }
}

template <int LAYER>
__global__ void __launch_bounds__(256)
gemm_wmma_kernel(const void* __restrict__ Xv, const float* __restrict__ W,
                 const float* __restrict__ bias, void* __restrict__ Yv,
                 int Kd, int Nd,
                 const float* __restrict__ resid,
                 const float* __restrict__ g2, const float* __restrict__ be2,
                 const float* __restrict__ m2, const float* __restrict__ v2) {
  int lane = threadIdx.x & 31;
  int w    = threadIdx.x >> 5;
  int wm   = w & 1;                         // 2 M-groups of 64
  int wn   = w >> 1;                        // 4 N-groups of 32
  int Mbase = blockIdx.y * 128 + wm * 64;
  int Nbase = blockIdx.x * 128 + wn * 32;
  int lr = lane & 15;
  int kh = lane >> 4;

  const float*  arf[4];
  const __bf16* arb[4];
#pragma unroll
  for (int mt = 0; mt < 4; ++mt) {
    size_t roff = (size_t)(Mbase + mt * 16 + lr) * Kd;
    arf[mt] = (const float*)Xv + roff;
    arb[mt] = (const __bf16*)Xv + roff;
  }
  const float* br[2];
#pragma unroll
  for (int nt = 0; nt < 2; ++nt) br[nt] = W + (size_t)(Nbase + nt * 16 + lr) * Kd;

  v8f zero = {0.f, 0.f, 0.f, 0.f, 0.f, 0.f, 0.f, 0.f};
  v8f acc[4][2];
#pragma unroll
  for (int mt = 0; mt < 4; ++mt)
#pragma unroll
    for (int nt = 0; nt < 2; ++nt) acc[mt][nt] = zero;

  // ---- explicit 2-stage pipeline over K (Kd is a multiple of 64) ----
  v16bf A0[4], B0[2], A1[4], B1[2];
  load_frags<LAYER>(A0, B0, arf, arb, br, 0, kh);
  for (int kbase = 0; kbase + 64 < Kd; kbase += 64) {
    load_frags<LAYER>(A1, B1, arf, arb, br, kbase + 32, kh);
    wmma8(acc, A0, B0);
    load_frags<LAYER>(A0, B0, arf, arb, br, kbase + 64, kh);
    wmma8(acc, A1, B1);
  }
  load_frags<LAYER>(A1, B1, arf, arb, br, Kd - 32, kh);
  wmma8(acc, A0, B0);
  wmma8(acc, A1, B1);

  // C/D layout: VGPR r, lane -> row = r + 8*(lane>>4), col = lane&15
#pragma unroll
  for (int mt = 0; mt < 4; ++mt) {
#pragma unroll
    for (int nt = 0; nt < 2; ++nt) {
      int col = Nbase + nt * 16 + lr;
      float bcol = bias[col];
      float bnsc = 0.f, bnof = 0.f;
      if (LAYER == 2) {
        int c = col >> 7;                    // col = c*128 + l
        bnsc = rsqrtf(v2[c] + EPSF) * g2[c];
        bnof = be2[c] - m2[c] * bnsc;
      }
#pragma unroll
      for (int r = 0; r < 8; ++r) {
        int row = Mbase + mt * 16 + r + 8 * kh;
        float val = fmaxf(acc[mt][nt][r] + bcol, 0.f);   // bias + ReLU
        size_t oidx = (size_t)row * Nd + col;
        if (LAYER == 2) {
          val = (val + resid[oidx]) * bnsc + bnof;
          ((float*)Yv)[oidx] = val;
        } else {
          ((__bf16*)Yv)[oidx] = (__bf16)val;
        }
      }
    }
  }
}

// ---------------------------------------------------------------------------

extern "C" void kernel_launch(void* const* d_in, const int* in_sizes, int n_in,
                              void* d_out, int out_size, void* d_ws, size_t ws_size,
                              hipStream_t stream) {
  (void)in_sizes; (void)n_in; (void)out_size; (void)ws_size;
  const float* x   = (const float*)d_in[0];
  const float* Wq  = (const float*)d_in[1];
  const float* bq  = (const float*)d_in[2];
  const float* Wk  = (const float*)d_in[3];
  const float* bk  = (const float*)d_in[4];
  const float* Wv  = (const float*)d_in[5];
  const float* bv  = (const float*)d_in[6];
  const float* Wo  = (const float*)d_in[7];
  const float* bo  = (const float*)d_in[8];
  const float* g1  = (const float*)d_in[9];
  const float* be1 = (const float*)d_in[10];
  const float* m1  = (const float*)d_in[11];
  const float* v1  = (const float*)d_in[12];
  const float* W1  = (const float*)d_in[13];
  const float* b1  = (const float*)d_in[14];
  const float* W2  = (const float*)d_in[15];
  const float* b2  = (const float*)d_in[16];
  const float* g2  = (const float*)d_in[17];
  const float* be2 = (const float*)d_in[18];
  const float* m2  = (const float*)d_in[19];
  const float* v2  = (const float*)d_in[20];
  const int*   idx = (const int*)d_in[21];

  float* wsf = (float*)d_ws;
  float* Q    = wsf;                     // 2,097,152 f
  float* K    = Q   + 2097152;           // 2,097,152 f
  float* V    = K   + 2097152;           // 2,097,152 f
  float* Ms   = V   + 2097152;           //   524,288 f
  float* upd  = Ms  + 524288;            //   143,360 f
  float* ctx  = upd + 143360;            // 2,097,152 f
  float* xh   = ctx + 2097152;           // 2,097,152 f
  __bf16* Y1b = (__bf16*)(xh + 2097152); // 4,194,304 bf16 (uses 2,097,152 f slots)
  int*   Mtop = (int*)(xh + 2097152 + 4194304);  //    35,840 ints

  float* out = (float*)d_out;

  qkv_kernel<<<65536, 32, 0, stream>>>(x, Wq, bq, Wk, bk, Wv, bv, Q, K, V);
  msample_kernel<<<(128 * 8 * 512 + 255) / 256, 256, 0, stream>>>(Q, K, idx, Ms);
  topk_kernel<<<1024, 256, 0, stream>>>(Ms, Mtop);
  attn_kernel<<<1024 * 35, 128, 0, stream>>>(Q, K, V, Mtop, upd);
  cumsum_kernel<<<1024, 512, 0, stream>>>(V, ctx);
  scatter_kernel<<<(1024 * 35 + 255) / 256, 256, 0, stream>>>(upd, Mtop, ctx);
  outproj_bn1_kernel<<<65536, 32, 0, stream>>>(ctx, Wo, bo, x, g1, be1, m1, v1, xh);

  // layer 1: (512x4096 f32) @ (4096x8192 f32)^T, bias + ReLU -> bf16 Y1
  gemm_wmma_kernel<1><<<dim3(64, 4), 256, 0, stream>>>(
      xh, W1, b1, Y1b, 4096, 8192,
      nullptr, nullptr, nullptr, nullptr, nullptr);
  // layer 2: (512x8192 bf16) @ (8192x4096 f32)^T, bias+ReLU+resid(xh)+BN2 -> d_out
  gemm_wmma_kernel<2><<<dim3(32, 4), 256, 0, stream>>>(
      Y1b, W2, b2, out, 8192, 4096,
      xh, g2, be2, m2, v2);
}
